// UniAttention_25125558682135
// MI455X (gfx1250) — matrix operand order repaired and checked by reading, not compiled, not run on hardware
//
#include <hip/hip_runtime.h>
#include <hip/hip_bf16.h>

// ---------------------------------------------------------------------------
// Causal MHA for MI455X (gfx1250, wave32, WMMA + TDM + async-LDS).
// Compute-bound (~103 GFLOP vs ~56MB traffic) -> f16 WMMA with f32 accum.
// ---------------------------------------------------------------------------

#define BATCH   2
#define SEQ     2048
#define HIDDEN  1024
#define HEADS   16
#define HSZ     64            // head size
#define NEG_INF (-10000.0f)

typedef __attribute__((ext_vector_type(16))) _Float16     v16h;
typedef __attribute__((ext_vector_type(8)))  float        v8f;
typedef __attribute__((ext_vector_type(4)))  unsigned int u32x4;
typedef __attribute__((ext_vector_type(8)))  unsigned int u32x8;

__device__ inline v8f wmma_f16(v16h a, v16h b, v8f c) {
  return __builtin_amdgcn_wmma_f32_16x16x32_f16(
      /*neg_a=*/false, a, /*neg_b=*/false, b,
      /*c_mod=*/(short)0, c, /*reuse_a=*/false, /*reuse_b=*/false);
}

// A-fragment: 16x32 f16 tile, row-major with row stride `ld` (halfs).
// Lane L: row m = L&15, half-group hi = L>>4.
//   a[0..7]  = A[m][hi*16 + i], a[8..15] = A[m][hi*16 + 8 + i]
// Two contiguous 16B runs per lane -> b128 loads.
__device__ inline v16h load_a_frag(const _Float16* p, int ld) {
  const int lane = threadIdx.x & 31;
  const int m = lane & 15, hi = lane >> 4;
  const _Float16* row = p + (size_t)m * ld;
  v16h a;
#pragma unroll
  for (int i = 0; i < 8; ++i) a[i] = row[hi * 16 + i];
#pragma unroll
  for (int i = 0; i < 8; ++i) a[8 + i] = row[hi * 16 + 8 + i];
  return a;
}

// B-fragment for C = A * W^T with W row-major [N x K] (nn.Linear layout).
// b[i] = W[n = L&15][ (L>>4)*16 + i ] -> one contiguous 32B run per lane.
__device__ inline v16h load_b_fragT(const _Float16* p, int ld) {
  const int lane = threadIdx.x & 31;
  const int n = lane & 15, hi = lane >> 4;
  const _Float16* row = p + (size_t)n * ld + hi * 16;
  v16h b;
#pragma unroll
  for (int i = 0; i < 16; ++i) b[i] = row[i];
  return b;
}

// B-fragment for row-major KxN operand (P*V path; served from LDS).
__device__ inline v16h load_b_frag(const _Float16* p, int ld) {
  const int lane = threadIdx.x & 31;
  const int n = lane & 15, hi = lane >> 4;
  const _Float16* col = p + n + (size_t)(hi * 16) * ld;
  v16h b;
#pragma unroll
  for (int i = 0; i < 16; ++i) b[i] = col[(size_t)i * ld];
  return b;
}

// ---------------------------------------------------------------------------
// Prep kernels
// ---------------------------------------------------------------------------
__global__ void k_prep_x(const float* __restrict__ x, const float* __restrict__ pos,
                         _Float16* __restrict__ xp, _Float16* __restrict__ xh,
                         int total, int ph) {
  int i = blockIdx.x * blockDim.x + threadIdx.x;
  if (i >= total) return;
  float xv = x[i];
  xh[i] = (_Float16)xv;
  xp[i] = (_Float16)(xv + pos[i % ph]);   // ph is a power of two -> AND
}

__global__ void k_cast(const float* __restrict__ w, _Float16* __restrict__ wh, int total) {
  int i = blockIdx.x * blockDim.x + threadIdx.x;
  if (i >= total) return;
  wh[i] = (_Float16)w[i];
}

// ---------------------------------------------------------------------------
// GEMM: C[m][n] = sum_k A[m][k] * W[n][k]   (A: MxK, W: NxK row-major)
// 256 threads = 8 waves; block tile 256x64; wave tile 32x64:
// per K-step 2 A-frags + 4 B-frags feed 8 WMMAs.
// ---------------------------------------------------------------------------
template <typename OutT>
__global__ __launch_bounds__(256)
void k_gemm(const _Float16* __restrict__ A, const _Float16* __restrict__ W,
            OutT* __restrict__ C, int M, int N, int K, int ldc) {
  const int w    = threadIdx.x >> 5;     // wave id (wave32)
  const int lane = threadIdx.x & 31;
  const int n    = lane & 15, hi = lane >> 4;
  const int row0 = blockIdx.x * 256 + w * 32;
  const int col0 = blockIdx.y * 64;
  if (row0 >= M || col0 >= N) return;

  v8f a00 = {}, a01 = {}, a02 = {}, a03 = {};
  v8f a10 = {}, a11 = {}, a12 = {}, a13 = {};
  const _Float16* ap0 = A + (size_t)row0 * K;
  const _Float16* ap1 = A + (size_t)(row0 + 16) * K;
  const _Float16* w0  = W + (size_t)col0 * K;

  for (int k = 0; k < K; k += 32) {
    __builtin_prefetch(ap0 + k + 128, 0, 0);     // global_prefetch_b8
    v16h fa0 = load_a_frag(ap0 + k, K);
    v16h fa1 = load_a_frag(ap1 + k, K);
    v16h fb;
    fb = load_b_fragT(w0 + k, K);
    a00 = wmma_f16(fa0, fb, a00);  a10 = wmma_f16(fa1, fb, a10);
    fb = load_b_fragT(w0 + (size_t)16 * K + k, K);
    a01 = wmma_f16(fa0, fb, a01);  a11 = wmma_f16(fa1, fb, a11);
    fb = load_b_fragT(w0 + (size_t)32 * K + k, K);
    a02 = wmma_f16(fa0, fb, a02);  a12 = wmma_f16(fa1, fb, a12);
    fb = load_b_fragT(w0 + (size_t)48 * K + k, K);
    a03 = wmma_f16(fa0, fb, a03);  a13 = wmma_f16(fa1, fb, a13);
  }

  OutT* cp0 = C + (size_t)row0 * ldc + col0;
  OutT* cp1 = C + (size_t)(row0 + 16) * ldc + col0;
#pragma unroll
  for (int r = 0; r < 8; ++r) {
    const int m = r + 8 * hi;
    cp0[(size_t)m * ldc + n]      = (OutT)a00[r];
    cp0[(size_t)m * ldc + 16 + n] = (OutT)a01[r];
    cp0[(size_t)m * ldc + 32 + n] = (OutT)a02[r];
    cp0[(size_t)m * ldc + 48 + n] = (OutT)a03[r];
    cp1[(size_t)m * ldc + n]      = (OutT)a10[r];
    cp1[(size_t)m * ldc + 16 + n] = (OutT)a11[r];
    cp1[(size_t)m * ldc + 32 + n] = (OutT)a12[r];
    cp1[(size_t)m * ldc + 48 + n] = (OutT)a13[r];
  }
}

// ---------------------------------------------------------------------------
// Flash attention: block = (b,h, 64-query tile), 4 waves x 16 query rows.
// K tile staged via TDM tensor_load_to_lds (TENSORcnt); V tile staged via
// global_load_async_to_lds_b128 (ASYNCcnt). Per 32-key tile:
// 4 WMMAs for S = Q*K^T, online softmax, 4 WMMAs for P*V.
// ---------------------------------------------------------------------------
__global__ __launch_bounds__(128)
void k_attention(const _Float16* __restrict__ qk, const _Float16* __restrict__ v,
                 _Float16* __restrict__ attn) {
  __shared__ __align__(16) _Float16 lds_k[4][32 * HSZ];  // K tiles (per wave)
  __shared__ __align__(16) _Float16 lds_v[4][32 * HSZ];  // V tiles (per wave)
  __shared__ __align__(16) _Float16 lds_p[4][16 * 32];   // P tiles (per wave)

  const int bh   = blockIdx.x;
  const int b    = bh / HEADS, h = bh % HEADS;
  const int w    = threadIdx.x >> 5;
  const int lane = threadIdx.x & 31;
  const int n    = lane & 15, hi = lane >> 4;
  const int qbase = blockIdx.y * 64 + w * 16;

  const int ldqk = 2 * HIDDEN;
  const _Float16* qp  = qk + ((size_t)b * SEQ + qbase) * ldqk + h * HSZ;
  const _Float16* kp0 = qk + (size_t)b * SEQ * ldqk + HIDDEN + h * HSZ;
  const _Float16* vp0 = v  + (size_t)b * SEQ * HIDDEN + h * HSZ;

  _Float16* Kt = lds_k[w];
  _Float16* Vt = lds_v[w];
  _Float16* Pt = lds_p[w];
  const unsigned kt_lds = (unsigned)(size_t)(void*)Kt;   // LDS byte offset (addr[31:0])
  const unsigned vt_lds = (unsigned)(size_t)(void*)Vt;

  // Q fragments for head-dim chunks [0,32) and [32,64) stay in registers.
  const v16h aq0 = load_a_frag(qp, ldqk);
  const v16h aq1 = load_a_frag(qp + 32, ldqk);

  v8f o0 = {}, o1 = {}, o2 = {}, o3 = {};
  float mrow[8], lrow[8];
#pragma unroll
  for (int r = 0; r < 8; ++r) { mrow[r] = -1e30f; lrow[r] = 0.0f; }

  const int nkt = (qbase + 16 + 31) >> 5;   // causal: key tiles of 32
  for (int kt = 0; kt < nkt; ++kt) {
    const int key0 = kt * 32;

    // --- TDM: K tile (32 keys x 64 halfs, row stride 2048 halfs) -> LDS ---
    {
      const unsigned long long ga =
          (unsigned long long)(size_t)(kp0 + (size_t)key0 * ldqk);
      u32x4 g0;
      g0[0] = 1u;                                   // count=1 (valid user D#)
      g0[1] = kt_lds;                               // lds_addr (bytes)
      g0[2] = (unsigned)(ga & 0xFFFFFFFFu);         // global_addr[31:0]
      g0[3] = (unsigned)((ga >> 32) & 0x01FFFFFFu)  // global_addr[56:32]
            | (2u << 30);                           // type=2 (image)
      u32x8 g1;
      g1[0] = (1u << 16);                           // wg_mask=0, data_size=2B
      g1[1] = (64u & 0xFFFFu) << 16;                // tensor_dim0[15:0] @bit48
      g1[2] = (64u >> 16) | ((32u & 0xFFFFu) << 16);// dim0 hi | tensor_dim1 lo
      g1[3] = (32u >> 16) | (64u << 16);            // dim1 hi | tile_dim0=64
      g1[4] = 32u;                                  // tile_dim1=32, tile_dim2=0
      g1[5] = 2048u;                                // tensor_dim0_stride lo32
      g1[6] = 0u;                                   // stride hi | dim1_stride lo
      g1[7] = 0u;
      asm volatile("tensor_load_to_lds %0, %1" :: "s"(g0), "s"(g1) : "memory");
    }

    // --- async: V tile (32 keys x 64 halfs) -> LDS, 16B per lane per issue ---
    {
      unsigned long long ga =
          (unsigned long long)(size_t)(vp0 + (size_t)(key0 + lane) * HIDDEN);
      unsigned ldst = vt_lds + lane * 128;          // one 128B key row per lane
#pragma unroll
      for (int c = 0; c < 8; ++c) {
        asm volatile("global_load_async_to_lds_b128 %0, %1, off"
                     :: "v"(ldst), "v"(ga) : "memory");
        ldst += 16; ga += 16;
      }
    }

    __builtin_amdgcn_s_wait_tensorcnt(0);           // K tile resident

    // S = Q * K^T for 16 rows x 32 keys (two 16-col fragments), from LDS.
    v8f s0 = {}, s1 = {};
    s0 = wmma_f16(aq0, load_b_fragT(Kt, HSZ), s0);
    s0 = wmma_f16(aq1, load_b_fragT(Kt + 32, HSZ), s0);
    s1 = wmma_f16(aq0, load_b_fragT(Kt + 16 * HSZ, HSZ), s1);
    s1 = wmma_f16(aq1, load_b_fragT(Kt + 16 * HSZ + 32, HSZ), s1);

    // Online softmax per row (row m = r + 8*hi lives in a 16-lane group).
#pragma unroll
    for (int r = 0; r < 8; ++r) {
      const int m    = r + 8 * hi;
      const int qrow = qbase + m;
      float v0 = s0[r] * 0.125f;             // 1/sqrt(64)
      float v1 = s1[r] * 0.125f;
      if (key0 + n      > qrow) v0 = NEG_INF;
      if (key0 + 16 + n > qrow) v1 = NEG_INF;

      float rmax = fmaxf(v0, v1);
#pragma unroll
      for (int off = 1; off < 16; off <<= 1)
        rmax = fmaxf(rmax, __shfl_xor(rmax, off, 32));
      const float mnew  = fmaxf(mrow[r], rmax);
      const float alpha = __expf(mrow[r] - mnew);
      const float p0 = __expf(v0 - mnew);
      const float p1 = __expf(v1 - mnew);
      float psum = p0 + p1;
#pragma unroll
      for (int off = 1; off < 16; off <<= 1)
        psum += __shfl_xor(psum, off, 32);
      lrow[r] = lrow[r] * alpha + psum;
      mrow[r] = mnew;
      o0[r] *= alpha; o1[r] *= alpha; o2[r] *= alpha; o3[r] *= alpha;

      // C-layout -> tile layout in LDS, re-read below as an A-fragment.
      Pt[m * 32 + n]      = (_Float16)p0;
      Pt[m * 32 + 16 + n] = (_Float16)p1;
    }

    asm volatile("s_wait_asynccnt 0x0" ::: "memory");  // V tile resident
    asm volatile("s_wait_dscnt 0x0" ::: "memory");     // P stores visible
    __builtin_amdgcn_wave_barrier();

    // O += P * V  (P: 16x32 A-frag from LDS; V: 32x64 in 4 col chunks).
    const v16h ap = load_a_frag(Pt, 32);
    o0 = wmma_f16(ap, load_b_frag(Vt + 0,  HSZ), o0);
    o1 = wmma_f16(ap, load_b_frag(Vt + 16, HSZ), o1);
    o2 = wmma_f16(ap, load_b_frag(Vt + 32, HSZ), o2);
    o3 = wmma_f16(ap, load_b_frag(Vt + 48, HSZ), o3);

    asm volatile("s_wait_dscnt 0x0" ::: "memory");     // LDS reads done before overwrite
    __builtin_amdgcn_wave_barrier();
  }

  // Epilogue: normalize and store f16 context [b, q, h*64 + f].
  _Float16* op = attn + ((size_t)b * SEQ + qbase) * HIDDEN + h * HSZ;
#pragma unroll
  for (int r = 0; r < 8; ++r) {
    const int m = r + 8 * hi;
    const float inv = 1.0f / lrow[r];
    op[(size_t)m * HIDDEN + n]      = (_Float16)(o0[r] * inv);
    op[(size_t)m * HIDDEN + 16 + n] = (_Float16)(o1[r] * inv);
    op[(size_t)m * HIDDEN + 32 + n] = (_Float16)(o2[r] * inv);
    op[(size_t)m * HIDDEN + 48 + n] = (_Float16)(o3[r] * inv);
  }
}

// ---------------------------------------------------------------------------
extern "C" void kernel_launch(void* const* d_in, const int* in_sizes, int n_in,
                              void* d_out, int out_size, void* d_ws, size_t ws_size,
                              hipStream_t stream) {
  const float* x    = (const float*)d_in[0];
  const float* pos  = (const float*)d_in[1];
  const float* Wqkv = (const float*)d_in[2];
  const float* Wout = (const float*)d_in[3];
  float* out = (float*)d_out;

  const size_t MROWS = (size_t)BATCH * SEQ;   // 4096 token rows

  // f16 workspace carve-out (~59 MB total).
  _Float16* ws = (_Float16*)d_ws;
  size_t off = 0;
  _Float16* xp_h   = ws + off; off += MROWS * HIDDEN;
  _Float16* x_h    = ws + off; off += MROWS * HIDDEN;
  _Float16* wqkv_h = ws + off; off += (size_t)3 * HIDDEN * HIDDEN;
  _Float16* wout_h = ws + off; off += (size_t)HIDDEN * HIDDEN;
  _Float16* qk_h   = ws + off; off += MROWS * 2 * HIDDEN;   // [row, q(1024)|k(1024)]
  _Float16* v_h    = ws + off; off += MROWS * HIDDEN;
  _Float16* attn_h = ws + off;

  const int totalx = (int)(MROWS * HIDDEN);
  k_prep_x<<<(totalx + 255) / 256, 256, 0, stream>>>(x, pos, xp_h, x_h, totalx, SEQ * HIDDEN);
  const int totw = 3 * HIDDEN * HIDDEN;
  k_cast<<<(totw + 255) / 256, 256, 0, stream>>>(Wqkv, wqkv_h, totw);
  const int totw2 = HIDDEN * HIDDEN;
  k_cast<<<(totw2 + 255) / 256, 256, 0, stream>>>(Wout, wout_h, totw2);

  // Q|K projection: [4096 x 1024] x [2048 x 1024]^T -> [4096 x 2048]
  dim3 gqk((unsigned)(MROWS / 256), (unsigned)(2 * HIDDEN / 64));
  k_gemm<_Float16><<<gqk, 256, 0, stream>>>(xp_h, wqkv_h, qk_h,
                                            (int)MROWS, 2 * HIDDEN, HIDDEN, 2 * HIDDEN);
  // V projection (raw x): [4096 x 1024] x [1024 x 1024]^T
  dim3 gv((unsigned)(MROWS / 256), (unsigned)(HIDDEN / 64));
  k_gemm<_Float16><<<gv, 256, 0, stream>>>(x_h, wqkv_h + (size_t)2 * HIDDEN * HIDDEN, v_h,
                                           (int)MROWS, HIDDEN, HIDDEN, HIDDEN);
  // Flash attention per (batch*head, 64-query tile)
  dim3 ga(BATCH * HEADS, SEQ / 64);
  k_attention<<<ga, 128, 0, stream>>>(qk_h, v_h, attn_h);

  // Output projection -> fp32 result
  dim3 go((unsigned)(MROWS / 256), (unsigned)(HIDDEN / 64));
  k_gemm<float><<<go, 256, 0, stream>>>(attn_h, wout_h, out,
                                        (int)MROWS, HIDDEN, HIDDEN, HIDDEN);
}
